// CharTaggerBiLSTM_39848706573850
// MI455X (gfx1250) — compile-verified, matrix operand order checked by reading, not run on hardware
//
#include <hip/hip_runtime.h>
#include <hip/hip_bf16.h>
#include <stdint.h>

typedef __attribute__((ext_vector_type(16))) __bf16 v16bf;
typedef __attribute__((ext_vector_type(8)))  float  v8f;
typedef unsigned short u16;

// ---------- scalar helpers (native bf16 converts) ----------
__device__ __forceinline__ u16 f2bf(float f) {
    __bf16 h = (__bf16)f;
    return __builtin_bit_cast(u16, h);
}
__device__ __forceinline__ float bf2f(u16 b) {
    __bf16 h = __builtin_bit_cast(__bf16, b);
    return (float)h;
}
__device__ __forceinline__ float sigf(float x) { return 1.0f / (1.0f + __expf(-x)); }

// Element offset inside a 16x32 bf16 WMMA operand tile (512 elems).
// A-matrix: first arg = m (row); B-matrix: first arg = n (col). Each lane owns
// 16 contiguous bf16 (32 bytes) -> one 32B load per operand.
__device__ __host__ __forceinline__ int aswz(int mn, int k) {
    int lane = mn + 16 * ((k >> 3) & 1);
    int idx  = (k & 7) + 8 * ((k >> 4) & 1);
    return lane * 16 + idx;
}

__device__ __forceinline__ v8f wmma_bf16(v16bf a, v16bf b, v8f c) {
    return __builtin_amdgcn_wmma_f32_16x16x32_bf16(false, a, false, b, (short)0, c, false, false);
}

// ---------------------------------------------------------------------------
// Prep 1: ig_table[c][g] = emb[c] . cW_ih[g] + cb_ih[g] + cb_hh[g]
// 100x1024 f32 (400KB, L2-resident) — folds char input projection + biases
// into a gather.
// ---------------------------------------------------------------------------
__global__ __launch_bounds__(256) void ig_table_kernel(
    const float* __restrict__ emb, const float* __restrict__ cWih,
    const float* __restrict__ cbih, const float* __restrict__ cbhh,
    float* __restrict__ ig)
{
    int idx = blockIdx.x * 256 + threadIdx.x;
    if (idx >= 100 * 1024) return;
    int ch = idx >> 10, g = idx & 1023;
    const float* e = emb  + ch * 64;
    const float* w = cWih + g * 64;
    float s = cbih[g] + cbhh[g];
#pragma unroll 8
    for (int k = 0; k < 64; ++k) s += e[k] * w[k];
    ig[idx] = s;
}

// ---------------------------------------------------------------------------
// Prep 2: swizzle row-major f32 W[n][k] (used as B = W^T in D = A*B) into
// bf16 WMMA B-operand tiles. Optional gate interleave: column n' = 4*j + gate
// maps to original row n = gate*Hu + j.
// ---------------------------------------------------------------------------
__global__ __launch_bounds__(256) void swizzle_kernel(
    const float* __restrict__ W, u16* __restrict__ out,
    int Ncols, int Nvalid, int Kw, int KTtot, int ktOff, int interleave, int Hu)
{
    int idx = blockIdx.x * 256 + threadIdx.x;
    if (idx >= Ncols * Kw) return;
    int np = idx / Kw, k = idx % Kw;
    int n;
    if (interleave) { int j = np >> 2, g = np & 3; n = g * Hu + j; }
    else            { n = np; }
    float v = (n < Nvalid) ? W[(size_t)n * Kw + k] : 0.0f;
    int nt = np >> 4, nl = np & 15, ktw = k >> 5, kl = k & 31;
    out[((size_t)(nt * KTtot + ktOff + ktw)) * 512 + aswz(nl, kl)] = f2bf(v);
}

// ---------------------------------------------------------------------------
// Char LSTM: block = 16 words, 8 waves. Per step: gates[16x1024] =
// ig_table gather (acc init) + hA[16x256] @ cWhh^T via WMMA bf16, then fused
// elementwise cell update. `last` tracked per word length.
// ---------------------------------------------------------------------------
__global__ __launch_bounds__(256) void char_lstm_kernel(
    const int* __restrict__ x, const float* __restrict__ ig,
    const u16* __restrict__ whhT, u16* __restrict__ char_out)
{
    __shared__ __align__(32) u16 hA[8 * 512];        // 16x256 bf16, A-swizzled
    __shared__ __align__(32) u16 gatesB[16 * 1024];  // bf16 gate staging
    __shared__ int chars[16];
    __shared__ int lens[16];

    const int t = threadIdx.x, lane = t & 31, wave = t >> 5;
    const int w0 = blockIdx.x * 16;
    const int m  = t >> 4;            // elementwise: word row
    const int jb = (t & 15) * 16;     // elementwise: unit base

    if (t < 16) {
        const int* xr = x + (w0 + t) * 20;
        int len = 0;
#pragma unroll
        for (int i = 0; i < 20; ++i) len += (xr[i] != 0);
        lens[t] = len;
    }
    for (int i = t; i < 8 * 512; i += 256) hA[i] = 0;

    float c[16], last[16];
#pragma unroll
    for (int i = 0; i < 16; ++i) { c[i] = 0.f; last[i] = 0.f; }
    __syncthreads();

    for (int step = 0; step < 20; ++step) {
        if (t < 16) chars[t] = x[(w0 + t) * 20 + step];
        __syncthreads();

        // WMMA phase: wave owns gate columns [wave*128, wave*128+128)
#pragma unroll 1
        for (int nn = 0; nn < 8; ++nn) {
            const int ntile = wave * 8 + nn;
            const int col = ntile * 16 + (lane & 15);
            v8f acc;
#pragma unroll
            for (int r = 0; r < 8; ++r) {
                int mr = r + 8 * (lane >> 4);
                acc[r] = ig[(size_t)chars[mr] * 1024 + col];
            }
#pragma unroll 1
            for (int kt = 0; kt < 8; ++kt) {
                v16bf a = *(const v16bf*)&hA[kt * 512 + lane * 16];
                v16bf b = *(const v16bf*)&whhT[((size_t)ntile * 8 + kt) * 512 + lane * 16];
                acc = wmma_bf16(a, b, acc);
            }
#pragma unroll
            for (int r = 0; r < 8; ++r) {
                int mr = r + 8 * (lane >> 4);
                gatesB[mr * 1024 + col] = f2bf(acc[r]);
            }
        }
        __syncthreads();

        // elementwise LSTM cell (gate order i,f,g,o)
#pragma unroll
        for (int uu = 0; uu < 16; ++uu) {
            int j = jb + uu;
            float gi = bf2f(gatesB[m * 1024 + j]);
            float gf = bf2f(gatesB[m * 1024 + 256 + j]);
            float gg = bf2f(gatesB[m * 1024 + 512 + j]);
            float go = bf2f(gatesB[m * 1024 + 768 + j]);
            float cn = sigf(gf) * c[uu] + sigf(gi) * tanhf(gg);
            float h  = sigf(go) * tanhf(cn);
            c[uu] = cn;
            if (lens[m] > step) last[uu] = h;
            hA[(j >> 5) * 512 + aswz(m, j & 31)] = f2bf(h);
        }
        __syncthreads();
    }
#pragma unroll
    for (int uu = 0; uu < 16; ++uu)
        char_out[(size_t)(w0 + m) * 256 + jb + uu] = f2bf(last[uu]);
}

// ---------------------------------------------------------------------------
// Word BiLSTM: 16 blocks = 2 dirs x 8 batch-tiles of 16. Fused K=768 GEMM
// ([x|h] @ [Wih|Whh]^T, gate-interleaved columns) -> two 1024-col passes.
// ---------------------------------------------------------------------------
__global__ __launch_bounds__(256) void word_lstm_kernel(
    const u16* __restrict__ char_out,
    const u16* __restrict__ fwT, const u16* __restrict__ bwT,
    const float* __restrict__ fb_ih, const float* __restrict__ fb_hh,
    const float* __restrict__ bb_ih, const float* __restrict__ bb_hh,
    u16* __restrict__ hs)
{
    __shared__ __align__(32) u16 A[24 * 512];        // 16x768 (x: k<256, h: k>=256)
    __shared__ __align__(32) u16 gatesB[16 * 1024];  // one pass of gate cols

    const int t = threadIdx.x, lane = t & 31, wave = t >> 5;
    const int dir = blockIdx.x >> 3, bt = blockIdx.x & 7;
    const int b0 = bt * 16;
    const u16*   wT  = dir ? bwT   : fwT;
    const float* bih = dir ? bb_ih : fb_ih;
    const float* bhh = dir ? bb_hh : fb_hh;
    const int m = t >> 4, ub = (t & 15) * 16;

    float c[32], hreg[32];
#pragma unroll
    for (int i = 0; i < 32; ++i) c[i] = 0.f;
    for (int i = t; i < 24 * 512; i += 256) A[i] = 0;
    __syncthreads();

    for (int step = 0; step < 128; ++step) {
        const int s = dir ? (127 - step) : step;

        // stage x_t (already bf16) into A ktiles 0..7
        {
            const int kb = ub;  // 16 k per thread
            const u16* src = char_out + ((size_t)(b0 + m) * 128 + s) * 256 + kb;
#pragma unroll
            for (int kk = 0; kk < 16; ++kk) {
                int k = kb + kk;
                A[(k >> 5) * 512 + aswz(m, k & 31)] = src[kk];
            }
        }
        __syncthreads();

#pragma unroll
        for (int pass = 0; pass < 2; ++pass) {
#pragma unroll 1
            for (int nn = 0; nn < 8; ++nn) {
                const int ntile = pass * 64 + wave * 8 + nn;
                const int col = ntile * 16 + (lane & 15); // interleaved: 4*j+gate
                const int j = col >> 2, g = col & 3;
                const float bias = bih[g * 512 + j] + bhh[g * 512 + j];
                v8f acc;
#pragma unroll
                for (int r = 0; r < 8; ++r) acc[r] = bias;
#pragma unroll 1
                for (int kt = 0; kt < 24; ++kt) {
                    v16bf a = *(const v16bf*)&A[kt * 512 + lane * 16];
                    v16bf b = *(const v16bf*)&wT[((size_t)ntile * 24 + kt) * 512 + lane * 16];
                    acc = wmma_bf16(a, b, acc);
                }
                const int lc = col - pass * 1024;
#pragma unroll
                for (int r = 0; r < 8; ++r) {
                    int mr = r + 8 * (lane >> 4);
                    gatesB[mr * 1024 + lc] = f2bf(acc[r]);
                }
            }
            __syncthreads();

            // elementwise for units [pass*256 + ub, +16)
#pragma unroll
            for (int uu = 0; uu < 16; ++uu) {
                const int jl = ub + uu;            // pass-local unit
                const int base = m * 1024 + jl * 4;
                float gi = bf2f(gatesB[base + 0]);
                float gf = bf2f(gatesB[base + 1]);
                float gg = bf2f(gatesB[base + 2]);
                float go = bf2f(gatesB[base + 3]);
                const int ci = pass * 16 + uu;
                float cn = sigf(gf) * c[ci] + sigf(gi) * tanhf(gg);
                c[ci] = cn;
                hreg[ci] = sigf(go) * tanhf(cn);
            }
            __syncthreads();
        }

        // commit h_new to recurrent A-tiles + hs output
        u16* hsrow = hs + (((size_t)(b0 + m) * 128 + s) * 1024 + dir * 512);
#pragma unroll
        for (int ci = 0; ci < 32; ++ci) {
            const int j = (ci >> 4) * 256 + ub + (ci & 15);
            const u16 hb = f2bf(hreg[ci]);
            const int k = 256 + j;
            A[(k >> 5) * 512 + aswz(m, k & 31)] = hb;
            hsrow[j] = hb;
        }
        __syncthreads();
    }
}

// ---------------------------------------------------------------------------
// Fused MLP + log_softmax: 16 rows/block; 1024->256 (relu) ->256 (relu)
// ->64(pad of 50) all via WMMA chained through LDS bf16 A-tiles.
// ---------------------------------------------------------------------------
__global__ __launch_bounds__(256) void mlp_kernel(
    const u16* __restrict__ hs,
    const u16* __restrict__ w1T, const float* __restrict__ b1,
    const u16* __restrict__ w2T, const float* __restrict__ b2,
    const u16* __restrict__ w3T, const float* __restrict__ b3,
    float* __restrict__ out)
{
    __shared__ __align__(32) u16 hsA[32 * 512];  // 16x1024
    __shared__ __align__(32) u16 h1A[8 * 512];   // 16x256
    __shared__ __align__(32) u16 h2A[8 * 512];   // 16x256
    __shared__ float logits[16 * 64];

    const int t = threadIdx.x, lane = t & 31, wave = t >> 5;
    const size_t r0 = (size_t)blockIdx.x * 16;
    const int m = t >> 4;

    // stage hs rows (bf16) into A tiles
    {
        const int kb = (t & 15) * 64;
        const u16* src = hs + (r0 + m) * 1024 + kb;
#pragma unroll
        for (int kk = 0; kk < 64; ++kk) {
            int k = kb + kk;
            hsA[(k >> 5) * 512 + aswz(m, k & 31)] = src[kk];
        }
    }
    __syncthreads();

    // GEMM1: N=256 (16 ntiles, 2/wave), K=1024 (32 ktiles), relu
#pragma unroll 1
    for (int nn = 0; nn < 2; ++nn) {
        const int ntile = wave * 2 + nn;
        const int col = ntile * 16 + (lane & 15);
        const float bb = b1[col];
        v8f acc;
#pragma unroll
        for (int r = 0; r < 8; ++r) acc[r] = bb;
#pragma unroll 1
        for (int kt = 0; kt < 32; ++kt) {
            v16bf a = *(const v16bf*)&hsA[kt * 512 + lane * 16];
            v16bf b = *(const v16bf*)&w1T[((size_t)ntile * 32 + kt) * 512 + lane * 16];
            acc = wmma_bf16(a, b, acc);
        }
#pragma unroll
        for (int r = 0; r < 8; ++r) {
            int mr = r + 8 * (lane >> 4);
            float v = acc[r]; v = v > 0.f ? v : 0.f;
            h1A[(col >> 5) * 512 + aswz(mr, col & 31)] = f2bf(v);
        }
    }
    __syncthreads();

    // GEMM2: N=256, K=256 (8 ktiles), relu
#pragma unroll 1
    for (int nn = 0; nn < 2; ++nn) {
        const int ntile = wave * 2 + nn;
        const int col = ntile * 16 + (lane & 15);
        const float bb = b2[col];
        v8f acc;
#pragma unroll
        for (int r = 0; r < 8; ++r) acc[r] = bb;
#pragma unroll 1
        for (int kt = 0; kt < 8; ++kt) {
            v16bf a = *(const v16bf*)&h1A[kt * 512 + lane * 16];
            v16bf b = *(const v16bf*)&w2T[((size_t)ntile * 8 + kt) * 512 + lane * 16];
            acc = wmma_bf16(a, b, acc);
        }
#pragma unroll
        for (int r = 0; r < 8; ++r) {
            int mr = r + 8 * (lane >> 4);
            float v = acc[r]; v = v > 0.f ? v : 0.f;
            h2A[(col >> 5) * 512 + aswz(mr, col & 31)] = f2bf(v);
        }
    }
    __syncthreads();

    // GEMM3: N=64 (waves 0..3), K=256
    if (wave < 4) {
        const int ntile = wave;
        const int col = ntile * 16 + (lane & 15);
        const float bb = (col < 50) ? b3[col] : 0.f;
        v8f acc;
#pragma unroll
        for (int r = 0; r < 8; ++r) acc[r] = bb;
#pragma unroll 1
        for (int kt = 0; kt < 8; ++kt) {
            v16bf a = *(const v16bf*)&h2A[kt * 512 + lane * 16];
            v16bf b = *(const v16bf*)&w3T[((size_t)ntile * 8 + kt) * 512 + lane * 16];
            acc = wmma_bf16(a, b, acc);
        }
#pragma unroll
        for (int r = 0; r < 8; ++r)
            logits[(r + 8 * (lane >> 4)) * 64 + col] = acc[r];
    }
    __syncthreads();

    // log_softmax over 50 classes, one row per thread
    if (t < 16) {
        float* orow = out + (r0 + t) * 50;
        float mx = -1e30f;
        for (int k = 0; k < 50; ++k) mx = fmaxf(mx, logits[t * 64 + k]);
        float sum = 0.f;
        for (int k = 0; k < 50; ++k) sum += __expf(logits[t * 64 + k] - mx);
        float ls = __logf(sum);
        for (int k = 0; k < 50; ++k) orow[k] = logits[t * 64 + k] - mx - ls;
    }
}

// ---------------------------------------------------------------------------
extern "C" void kernel_launch(void* const* d_in, const int* in_sizes, int n_in,
                              void* d_out, int out_size, void* d_ws, size_t ws_size,
                              hipStream_t stream)
{
    const int*   x     = (const int*)  d_in[0];
    const float* emb   = (const float*)d_in[1];
    const float* cWih  = (const float*)d_in[2];
    const float* cWhh  = (const float*)d_in[3];
    const float* cbih  = (const float*)d_in[4];
    const float* cbhh  = (const float*)d_in[5];
    const float* fWih  = (const float*)d_in[6];
    const float* fWhh  = (const float*)d_in[7];
    const float* fbih  = (const float*)d_in[8];
    const float* fbhh  = (const float*)d_in[9];
    const float* bWih  = (const float*)d_in[10];
    const float* bWhh  = (const float*)d_in[11];
    const float* bbih  = (const float*)d_in[12];
    const float* bbhh  = (const float*)d_in[13];
    const float* W1    = (const float*)d_in[14];
    const float* b1    = (const float*)d_in[15];
    const float* W2    = (const float*)d_in[16];
    const float* b2    = (const float*)d_in[17];
    const float* W3    = (const float*)d_in[18];
    const float* b3    = (const float*)d_in[19];
    float* outp = (float*)d_out;

    char* ws = (char*)d_ws;
    size_t off = 0;
    auto carve = [&](size_t bytes) { size_t o = off; off = (off + bytes + 255) & ~(size_t)255; return o; };

    float* ig      = (float*)(ws + carve(100 * 1024 * 4));             // 400KB
    u16*   cwhhT   = (u16*)  (ws + carve((size_t)64 * 8  * 512 * 2));  // 512KB
    u16*   fwT     = (u16*)  (ws + carve((size_t)128 * 24 * 512 * 2)); // 3MB
    u16*   bwT     = (u16*)  (ws + carve((size_t)128 * 24 * 512 * 2)); // 3MB
    u16*   w1T     = (u16*)  (ws + carve((size_t)16 * 32 * 512 * 2));
    u16*   w2T     = (u16*)  (ws + carve((size_t)16 * 8  * 512 * 2));
    u16*   w3T     = (u16*)  (ws + carve((size_t)4  * 8  * 512 * 2));
    u16*   charout = (u16*)  (ws + carve((size_t)16384 * 256 * 2));    // 8MB
    u16*   hsb     = (u16*)  (ws + carve((size_t)16384 * 1024 * 2));   // 32MB
    (void)ws_size; (void)in_sizes; (void)n_in; (void)out_size;

    // ---- prep ----
    ig_table_kernel<<<(100 * 1024 + 255) / 256, 256, 0, stream>>>(emb, cWih, cbih, cbhh, ig);
    // char W_hh: [1024 x 256], no interleave
    swizzle_kernel<<<(1024 * 256 + 255) / 256, 256, 0, stream>>>(cWhh, cwhhT, 1024, 1024, 256, 8, 0, 0, 0);
    // word fwd: combined [Wih(K=256) | Whh(K=512)] tiles, gate-interleaved cols
    swizzle_kernel<<<(2048 * 256 + 255) / 256, 256, 0, stream>>>(fWih, fwT, 2048, 2048, 256, 24, 0, 1, 512);
    swizzle_kernel<<<(2048 * 512 + 255) / 256, 256, 0, stream>>>(fWhh, fwT, 2048, 2048, 512, 24, 8, 1, 512);
    swizzle_kernel<<<(2048 * 256 + 255) / 256, 256, 0, stream>>>(bWih, bwT, 2048, 2048, 256, 24, 0, 1, 512);
    swizzle_kernel<<<(2048 * 512 + 255) / 256, 256, 0, stream>>>(bWhh, bwT, 2048, 2048, 512, 24, 8, 1, 512);
    // MLP weights
    swizzle_kernel<<<(256 * 1024 + 255) / 256, 256, 0, stream>>>(W1, w1T, 256, 256, 1024, 32, 0, 0, 0);
    swizzle_kernel<<<(256 * 256 + 255) / 256, 256, 0, stream>>>(W2, w2T, 256, 256, 256, 8, 0, 0, 0);
    swizzle_kernel<<<(64 * 256 + 255) / 256, 256, 0, stream>>>(W3, w3T, 64, 50, 256, 8, 0, 0, 0);

    // ---- compute ----
    char_lstm_kernel<<<1024, 256, 0, stream>>>(x, ig, cwhhT, charout);
    word_lstm_kernel<<<16, 256, 0, stream>>>(charout, fwT, bwT, fbih, fbhh, bbih, bbhh, hsb);
    mlp_kernel<<<1024, 256, 0, stream>>>(hsb, w1T, b1, w2T, b2, w3T, b3, outp);
}
